// LongConv1d_66082366816305
// MI455X (gfx1250) — compile-verified, matrix operand order checked by reading, not run on hardware
//
#include <hip/hip_runtime.h>

// ---------------------------------------------------------------------------
// LongConv1d via FFT on MI455X (gfx1250, wave32, WMMA)
//   X:(8,64,65536) f32, filters:(128,64,1023) f32 -> out:(8,128,65536) f32
//   t=16 decimation split -> 128 signals/channel of 4096, rFFT, per-bin
//   complex GEMM (128x64 @ 64x128, 2049 bins) in bf16 WMMA w/ f32 accum,
//   irFFT + abs + log1p + InstanceNorm.
//   GEMM stages A/B tiles into LDS with GLOBAL_LOAD_ASYNC_TO_LDS_B128
//   (ASYNCcnt) -- the CDNA5 direct global->LDS path, no VGPR round trip.
// ---------------------------------------------------------------------------

#define NFFT   4096
#define LOG2N  12
#define NBINS  2049
#define BDIM   8
#define CDIM   64
#define HDIM   128
#define TDEC   16
#define TBDIM  128          // TDEC*BDIM
#define SIGU   8192         // TBDIM*CDIM
#define SIGV   8192         // HDIM*CDIM
#define BH     16384        // TBDIM*HDIM
#define NTOT   65536

typedef __attribute__((ext_vector_type(16))) __bf16        v16bf;
typedef __attribute__((ext_vector_type(16))) unsigned short v16us;
typedef __attribute__((ext_vector_type(8)))  float          v8f;

// round-to-nearest-even f32 -> bf16 (as raw ushort)
__device__ inline unsigned short f2bf(float f) {
  unsigned int u = __float_as_uint(f);
  u += 0x7FFFu + ((u >> 16) & 1u);
  return (unsigned short)(u >> 16);
}

// 4096-pt Stockham radix-2 FFT in LDS (autosort, natural order in/out).
// sign = -1 forward, +1 inverse (caller scales by 1/N for inverse).
// Result ends in (ar, ai) after the even number (12) of stages.
__device__ inline void fft4096_lds(float* ar, float* ai, float* br, float* bi,
                                   int tid, int nthr, float sign) {
  float* sr = ar; float* si = ai; float* dr = br; float* di = bi;
  for (int s = 0; s < LOG2N; ++s) {
    int Ns = 1 << s;
    __syncthreads();
    for (int j = tid; j < NFFT / 2; j += nthr) {
      int q = j & (Ns - 1);
      float ang = sign * 6.28318530717958647f * (float)q / (float)(Ns << 1);
      float wi, wr;
      __sincosf(ang, &wi, &wr);
      float v0r = sr[j],            v0i = si[j];
      float v1r = sr[j + NFFT / 2], v1i = si[j + NFFT / 2];
      float tr = v1r * wr - v1i * wi;
      float ti = v1r * wi + v1i * wr;
      int d = ((j >> s) << (s + 1)) + q;
      dr[d]      = v0r + tr;  di[d]      = v0i + ti;
      dr[d + Ns] = v0r - tr;  di[d + Ns] = v0i - ti;
    }
    float* t;
    t = sr; sr = dr; dr = t;
    t = si; si = di; di = t;
  }
  __syncthreads();
}

// --------------------------- forward FFT of X -------------------------------
// block = one signal (tb, c); u[tb,c,i] = X[b, c, i*16 + ti], tb = ti*8 + b
__global__ void __launch_bounds__(256)
k_fft_x(const float* __restrict__ X,
        unsigned short* __restrict__ Ur, unsigned short* __restrict__ Ui) {
  __shared__ float ar[NFFT], ai[NFFT], br[NFFT], bi[NFFT];
  int sig = blockIdx.x;               // tb*64 + c
  int tb = sig >> 6, c = sig & 63;
  int ti = tb >> 3, b0 = tb & 7;
  const float* xp = X + ((size_t)(b0 * CDIM + c)) * NTOT + ti;
  for (int i = threadIdx.x; i < NFFT; i += 256) {
    ar[i] = xp[(size_t)i * TDEC];
    ai[i] = 0.f;
  }
  fft4096_lds(ar, ai, br, bi, threadIdx.x, 256, -1.f);
  size_t o = (size_t)sig * NBINS;
  for (int n = threadIdx.x; n < NBINS; n += 256) {
    Ur[o + n] = f2bf(ar[n]);
    Ui[o + n] = f2bf(ai[n]);
  }
}

// ----------------------- forward FFT of filters (conj) ----------------------
__global__ void __launch_bounds__(256)
k_fft_filt(const float* __restrict__ F,
           unsigned short* __restrict__ Vr, unsigned short* __restrict__ Vi) {
  __shared__ float ar[NFFT], ai[NFFT], br[NFFT], bi[NFFT];
  int sig = blockIdx.x;               // h*64 + c
  const float* fp = F + (size_t)sig * 1023;
  for (int i = threadIdx.x; i < NFFT; i += 256) {
    ar[i] = (i < 1023) ? fp[i] : 0.f;
    ai[i] = 0.f;
  }
  fft4096_lds(ar, ai, br, bi, threadIdx.x, 256, -1.f);
  size_t o = (size_t)sig * NBINS;
  for (int n = threadIdx.x; n < NBINS; n += 256) {
    Vr[o + n] = f2bf(ar[n]);
    Vi[o + n] = f2bf(-ai[n]);         // conjugate
  }
}

// --------------------------- tiled transposes -------------------------------
__global__ void __launch_bounds__(256)
k_tr16(const unsigned short* __restrict__ src, unsigned short* __restrict__ dst,
       int R, int Cc) {
  __shared__ unsigned short t[32][33];
  int bx = blockIdx.x * 32, by = blockIdx.y * 32;
  int tx = threadIdx.x & 31, ty0 = threadIdx.x >> 5;
  for (int ty = ty0; ty < 32; ty += 8) {
    int r = by + ty, cc = bx + tx;
    if (r < R && cc < Cc) t[ty][tx] = src[(size_t)r * Cc + cc];
  }
  __syncthreads();
  for (int ty = ty0; ty < 32; ty += 8) {
    int r = bx + ty, cc = by + tx;
    if (r < Cc && cc < R) dst[(size_t)r * R + cc] = t[tx][ty];
  }
}

__global__ void __launch_bounds__(256)
k_trf(const float* __restrict__ src, float* __restrict__ dst, int R, int Cc) {
  __shared__ float t[32][33];
  int bx = blockIdx.x * 32, by = blockIdx.y * 32;
  int tx = threadIdx.x & 31, ty0 = threadIdx.x >> 5;
  for (int ty = ty0; ty < 32; ty += 8) {
    int r = by + ty, cc = bx + tx;
    if (r < R && cc < Cc) t[ty][tx] = src[(size_t)r * Cc + cc];
  }
  __syncthreads();
  for (int ty = ty0; ty < 32; ty += 8) {
    int r = bx + ty, cc = by + tx;
    if (r < Cc && cc < R) dst[(size_t)r * R + cc] = t[tx][ty];
  }
}

// --------------------- per-bin complex GEMM via WMMA ------------------------
// One block per bin n. A = u_f[n]: 128(tb) x 64(c) (bf16 re/im),
// B = conj(v_f)[n] stored as V[h][c]: output col h, K = c.
// y = u * V (V already conjugated):
//   Yr = Ar*Br - Ai*Bi ; Yi = Ar*Bi + Ai*Br
// LDS staging uses GLOBAL_LOAD_ASYNC_TO_LDS_B128 (GVS addressing) so data
// flows global->LDS without touching VGPRs; completion via s_wait_asynccnt.
// Dynamic LDS (no static LDS in this kernel) => LDS byte offset of smem == 0.
__global__ void __launch_bounds__(512)
k_gemm(const unsigned short* __restrict__ UrT, const unsigned short* __restrict__ UiT,
       const unsigned short* __restrict__ VrT, const unsigned short* __restrict__ ViT,
       float* __restrict__ Yr, float* __restrict__ Yi) {
  extern __shared__ unsigned short smem[];        // 64 KB dynamic LDS
  unsigned short* Ash0 = smem;                    // [128*64] re(u)
  unsigned short* Ash1 = smem + 8192;             // im(u)
  unsigned short* Bsh0 = smem + 16384;            // re(V)
  unsigned short* Bsh1 = smem + 24576;            // im(V)

  int n = blockIdx.x;
  size_t gbyte = (size_t)n * (8192 * sizeof(unsigned short)); // bin plane offset
  unsigned long long aR = (unsigned long long)((const char*)UrT + gbyte);
  unsigned long long aI = (unsigned long long)((const char*)UiT + gbyte);
  unsigned long long bR = (unsigned long long)((const char*)VrT + gbyte);
  unsigned long long bI = (unsigned long long)((const char*)ViT + gbyte);

  // 16 KB per array = 1024 chunks of 16B; 512 threads -> 2 chunks each.
#pragma unroll
  for (int c = 0; c < 2; ++c) {
    unsigned off = (threadIdx.x + c * 512) * 16u;  // byte offset in plane
    asm volatile("global_load_async_to_lds_b128 %0, %1, %2"
                 :: "v"(off + 0u),      "v"(off), "s"(aR) : "memory");
    asm volatile("global_load_async_to_lds_b128 %0, %1, %2"
                 :: "v"(off + 16384u),  "v"(off), "s"(aI) : "memory");
    asm volatile("global_load_async_to_lds_b128 %0, %1, %2"
                 :: "v"(off + 32768u),  "v"(off), "s"(bR) : "memory");
    asm volatile("global_load_async_to_lds_b128 %0, %1, %2"
                 :: "v"(off + 49152u),  "v"(off), "s"(bI) : "memory");
  }
  asm volatile("s_wait_asynccnt 0" ::: "memory");
  __syncthreads();

  int wave = threadIdx.x >> 5, lane = threadIdx.x & 31;
  int rBase = (wave >> 2) * 32;     // 4 row-blocks of 32
  int cBase = (wave & 3) * 32;      // 4 col-blocks of 32
  int lm = lane & 15;
  int lh = (lane >> 4) & 1;

  v8f accR[2][2] = {};
  v8f accI[2][2] = {};

#pragma unroll
  for (int kb = 0; kb < 64; kb += 32) {
    v16us fAR[2], fAI[2], fBR[2], fBI[2];
#pragma unroll
    for (int t = 0; t < 2; ++t) {
      int M  = rBase + t * 16 + lm;
      int Nc = cBase + t * 16 + lm;
#pragma unroll
      for (int e = 0; e < 16; ++e) {
        // bf16 A-fragment K mapping: VGPR0-3 -> K 0..7 (+8 for hi lanes),
        // VGPR4-7 -> K 16..23 (+8 for hi lanes)
        int K = kb + (e & 7) + ((e & 8) ? 16 : 0) + (lh ? 8 : 0);
        fAR[t][e] = Ash0[M * 64 + K];
        fAI[t][e] = Ash1[M * 64 + K];
        fBR[t][e] = Bsh0[Nc * 64 + K];
        fBI[t][e] = Bsh1[Nc * 64 + K];
      }
    }
    v16us fAIn[2];
    fAIn[0] = fAI[0] ^ (unsigned short)0x8000u;   // -Ai (bf16 sign flip)
    fAIn[1] = fAI[1] ^ (unsigned short)0x8000u;
#pragma unroll
    for (int t = 0; t < 2; ++t) {
      v16bf ar_  = __builtin_bit_cast(v16bf, fAR[t]);
      v16bf ai_  = __builtin_bit_cast(v16bf, fAI[t]);
      v16bf ain_ = __builtin_bit_cast(v16bf, fAIn[t]);
#pragma unroll
      for (int u = 0; u < 2; ++u) {
        v16bf br_ = __builtin_bit_cast(v16bf, fBR[u]);
        v16bf bi_ = __builtin_bit_cast(v16bf, fBI[u]);
        // Yr += Ar*Br - Ai*Bi   ;   Yi += Ar*Bi + Ai*Br
        accR[t][u] = __builtin_amdgcn_wmma_f32_16x16x32_bf16(
            false, ar_,  false, br_, (short)0, accR[t][u], false, false);
        accR[t][u] = __builtin_amdgcn_wmma_f32_16x16x32_bf16(
            false, ain_, false, bi_, (short)0, accR[t][u], false, false);
        accI[t][u] = __builtin_amdgcn_wmma_f32_16x16x32_bf16(
            false, ar_,  false, bi_, (short)0, accI[t][u], false, false);
        accI[t][u] = __builtin_amdgcn_wmma_f32_16x16x32_bf16(
            false, ai_,  false, br_, (short)0, accI[t][u], false, false);
      }
    }
  }

  size_t yo = (size_t)n * BH;
#pragma unroll
  for (int t = 0; t < 2; ++t)
#pragma unroll
    for (int u = 0; u < 2; ++u)
#pragma unroll
      for (int v = 0; v < 8; ++v) {
        int M  = rBase + t * 16 + v + lh * 8;
        int Nc = cBase + u * 16 + lm;
        Yr[yo + (size_t)M * HDIM + Nc] = accR[t][u][v];
        Yi[yo + (size_t)M * HDIM + Nc] = accI[t][u][v];
      }
}

// ------------- inverse FFT + abs + log1p + InstanceNorm + scatter -----------
__global__ void __launch_bounds__(256)
k_irfft_norm(const float* __restrict__ YrT, const float* __restrict__ YiT,
             float* __restrict__ out) {
  __shared__ float ar[NFFT], ai[NFFT], br[NFFT], bi[NFFT];
  int sig = blockIdx.x;               // tb*128 + h
  int tb = sig >> 7, h = sig & 127;
  int ti = tb >> 3, b0 = tb & 7;
  size_t o = (size_t)sig * NBINS;
  for (int i = threadIdx.x; i < NBINS; i += 256) {
    ar[i] = YrT[o + i];
    ai[i] = YiT[o + i];
  }
  __syncthreads();
  // Hermitian mirror for real inverse transform
  for (int i = NBINS + threadIdx.x; i < NFFT; i += 256) {
    ar[i] =  ar[NFFT - i];
    ai[i] = -ai[NFFT - i];
  }
  fft4096_lds(ar, ai, br, bi, threadIdx.x, 256, +1.f);

  float s = 0.f, s2 = 0.f;
  for (int i = threadIdx.x; i < NFFT; i += 256) {
    float v = fabsf(ar[i] * (1.f / NFFT)); // abs(irfft); relu is identity here
    v = __logf(1.f + v);                   // NLReLU
    ar[i] = v;
    s  += v;
    s2 += v * v;
  }
  br[threadIdx.x] = s;
  bi[threadIdx.x] = s2;
  __syncthreads();
  for (int off = 128; off > 0; off >>= 1) {
    if (threadIdx.x < off) {
      br[threadIdx.x] += br[threadIdx.x + off];
      bi[threadIdx.x] += bi[threadIdx.x + off];
    }
    __syncthreads();
  }
  float mean = br[0] * (1.f / NFFT);
  float var  = bi[0] * (1.f / NFFT) - mean * mean;
  float rs   = __frsqrt_rn(var + 1e-5f);

  float* op = out + ((size_t)b0 * HDIM + h) * NTOT + ti;
  for (int i = threadIdx.x; i < NFFT; i += 256) {
    op[(size_t)i * TDEC] = (ar[i] - mean) * rs;
  }
}

// ---------------------------------------------------------------------------
extern "C" void kernel_launch(void* const* d_in, const int* in_sizes, int n_in,
                              void* d_out, int out_size, void* d_ws, size_t ws_size,
                              hipStream_t stream) {
  const float* X = (const float*)d_in[0];      // (8,64,65536)
  const float* F = (const float*)d_in[1];      // (128,64,1023)
  float* out = (float*)d_out;                  // (8,128,65536)
  char* ws = (char*)d_ws;

  const size_t szBF = (size_t)SIGU * NBINS * sizeof(unsigned short); // 33.57 MB
  const size_t szF  = (size_t)NBINS * BH * sizeof(float);            // 134.3 MB (= 4*szBF)

  unsigned short* Ur  = (unsigned short*)(ws + 0 * szBF);
  unsigned short* Ui  = (unsigned short*)(ws + 1 * szBF);
  unsigned short* Vr  = (unsigned short*)(ws + 2 * szBF);
  unsigned short* Vi  = (unsigned short*)(ws + 3 * szBF);
  unsigned short* UrT = (unsigned short*)(ws + 4 * szBF);
  unsigned short* UiT = (unsigned short*)(ws + 5 * szBF);
  unsigned short* VrT = (unsigned short*)(ws + 6 * szBF);
  unsigned short* ViT = (unsigned short*)(ws + 7 * szBF);
  float* Yr = (float*)(ws + 8 * szBF);
  float* Yi = (float*)(ws + 8 * szBF + szF);
  // After the GEMM, the bf16 spectra are dead: reuse their space for y_f^T.
  float* YrT = (float*)(ws);
  float* YiT = (float*)(ws + szF);

  // 1) forward FFTs -> bf16 spectra, signal-major
  k_fft_x   <<<SIGU, 256, 0, stream>>>(X, Ur, Ui);
  k_fft_filt<<<SIGV, 256, 0, stream>>>(F, Vr, Vi);

  // 2) transpose to bin-major planes [n][row*64+c]
  dim3 gU((NBINS + 31) / 32, (SIGU + 31) / 32);
  k_tr16<<<gU, 256, 0, stream>>>(Ur, UrT, SIGU, NBINS);
  k_tr16<<<gU, 256, 0, stream>>>(Ui, UiT, SIGU, NBINS);
  k_tr16<<<gU, 256, 0, stream>>>(Vr, VrT, SIGV, NBINS);
  k_tr16<<<gU, 256, 0, stream>>>(Vi, ViT, SIGV, NBINS);

  // 3) per-bin complex GEMM (WMMA bf16 -> f32), async LDS staging, 64KB dyn LDS
  k_gemm<<<NBINS, 512, 65536, stream>>>(UrT, UiT, VrT, ViT, Yr, Yi);

  // 4) transpose y_f to signal-major [tb*128+h][n]
  dim3 gY((BH + 31) / 32, (NBINS + 31) / 32);
  k_trf<<<gY, 256, 0, stream>>>(Yr, YrT, NBINS, BH);
  k_trf<<<gY, 256, 0, stream>>>(Yi, YiT, NBINS, BH);

  // 5) inverse FFT + pointwise + InstanceNorm + interleaved scatter to out
  k_irfft_norm<<<BH, 256, 0, stream>>>(YrT, YiT, out);
}
